// CUDAClusteringModule_77163382440548
// MI455X (gfx1250) — compile-verified, matrix operand order checked by reading, not run on hardware
//
#include <hip/hip_runtime.h>
#include <hip/hip_bf16.h>

typedef __attribute__((ext_vector_type(16))) _Float16 v16h;
typedef __attribute__((ext_vector_type(8)))  _Float16 v8h;
typedef __attribute__((ext_vector_type(8)))  float    v8f;

#define NPTS        1000000
#define MAXV        8192
#define MAXP        32
#define MINPTS      10
#define SENTINEL    8192      // label sentinel (== MAXV in reference)
#define INVALID_SQ  1e30f
#define BIGF16      60000.0f  // f16-representable "infinity" for invalid voxels
#define PROP_ITERS  64
#define NSPLIT      4         // j-range splits per i-tile (4x wave parallelism)

// ---------------------------------------------------------------------------
// K0: workspace init (ws is poisoned; re-init everything each call)
// ---------------------------------------------------------------------------
__global__ void k_init(float4* __restrict__ C4, float4* __restrict__ voxsum,
                       float4* __restrict__ csum, int* __restrict__ cnt_cells,
                       int* __restrict__ cellToVid, int* __restrict__ fill,
                       int* __restrict__ ccnt) {
  int v = blockIdx.x * blockDim.x + threadIdx.x;
  if (v >= MAXV) return;
  C4[v]     = make_float4(0.f, 0.f, 0.f, INVALID_SQ);   // default: invalid voxel
  voxsum[v] = make_float4(0.f, 0.f, 0.f, 0.f);
  csum[v]   = make_float4(0.f, 0.f, 0.f, 0.f);
  cnt_cells[v] = 0;
  cellToVid[v] = 0;
  fill[v]      = 0;
  ccnt[v]      = 0;
}

// ---------------------------------------------------------------------------
// K1: per-point voxel-cell histogram (dense 20^3 grid; lex cell order ==
// jnp.unique sorted key order)
// ---------------------------------------------------------------------------
__device__ __forceinline__ int point_cell(float4 p) {
  int cx = min(max((int)floorf(p.x * 2.0f), 0), 19);
  int cy = min(max((int)floorf(p.y * 2.0f), 0), 19);
  int cz = min(max((int)floorf(p.z * 2.0f), 0), 19);
  return (cx * 20 + cy) * 20 + cz;
}

__global__ void k_count(const float4* __restrict__ pts, int* __restrict__ cnt_cells) {
  int i = blockIdx.x * blockDim.x + threadIdx.x;
  if (i >= NPTS) return;
  atomicAdd(&cnt_cells[point_cell(pts[i])], 1);
}

// ---------------------------------------------------------------------------
// K2: compact occupied cells -> voxel ids (scan over 8192 cells, one block)
// ---------------------------------------------------------------------------
__global__ void k_scan(const int* __restrict__ cnt_cells,
                       int* __restrict__ cellToVid, float4* __restrict__ C4) {
  __shared__ int sh[1024];
  const int t = threadIdx.x;
  const int base = t * 8;
  int occ[8];
  int sum = 0;
  #pragma unroll
  for (int k = 0; k < 8; ++k) {
    occ[k] = (cnt_cells[base + k] > 0) ? 1 : 0;
    sum += occ[k];
  }
  sh[t] = sum;
  __syncthreads();
  for (int off = 1; off < 1024; off <<= 1) {      // inclusive Hillis-Steele
    int v = (t >= off) ? sh[t - off] : 0;
    __syncthreads();
    sh[t] += v;
    __syncthreads();
  }
  int vid = sh[t] - sum;                           // exclusive prefix
  #pragma unroll
  for (int k = 0; k < 8; ++k) {
    int cell = base + k;
    if (occ[k]) {
      cellToVid[cell] = vid;
      int c = cnt_cells[cell];
      if (c >= MINPTS) {
        float fx = (float)(cell / 400);
        float fy = (float)((cell / 20) % 20);
        float fz = (float)(cell % 20);
        C4[vid] = make_float4(fx, fy, fz, fx * fx + fy * fy + fz * fz);
      }
      ++vid;
    }
  }
}

// ---------------------------------------------------------------------------
// K3: accumulate xyz sums per voxel, capped at MAXP points per voxel
// ---------------------------------------------------------------------------
__global__ void k_accum(const float4* __restrict__ pts,
                        const int* __restrict__ cellToVid,
                        int* __restrict__ fill, float* __restrict__ voxsum) {
  int i = blockIdx.x * blockDim.x + threadIdx.x;
  if (i >= NPTS) return;
  float4 p = pts[i];
  int vid = cellToVid[point_cell(p)];
  int slot = atomicAdd(&fill[vid], 1);
  if (slot < MAXP) {
    atomicAdd(&voxsum[4 * vid + 0], p.x);
    atomicAdd(&voxsum[4 * vid + 1], p.y);
    atomicAdd(&voxsum[4 * vid + 2], p.z);
  }
}

// ---------------------------------------------------------------------------
// K4: build pre-packed f16 WMMA fragments so the prop kernel's WMMA emits d2
// directly:  A_i = (-2x, -2y, -2z, sq_i, 1, 0...)   B_j = (x, y, z, 1, sq_j, 0...)
//            d2(i,j) = A_i . B_j = sq_i + sq_j - 2*Ci.Cj   (exact in f16/f32)
// Invalid voxels: coords=0, sq=60000 -> d2 > 4 against anything.
// Also init labels and the zero-pad fragment entries.
// ---------------------------------------------------------------------------
__global__ void k_prep(const float4* __restrict__ C4, v8h* __restrict__ Af16,
                       v8h* __restrict__ Bf16, int* __restrict__ lab) {
  int v = blockIdx.x * blockDim.x + threadIdx.x;
  if (v >= MAXV) return;
  float4 c = C4[v];
  bool valid = (c.w < 1e29f);
  float sq = valid ? c.w : BIGF16;
  v8h a, b;
  a[0] = (_Float16)(-2.0f * c.x);
  a[1] = (_Float16)(-2.0f * c.y);
  a[2] = (_Float16)(-2.0f * c.z);
  a[3] = (_Float16)sq;
  a[4] = (_Float16)1.0f;
  a[5] = (_Float16)0.0f; a[6] = (_Float16)0.0f; a[7] = (_Float16)0.0f;
  b[0] = (_Float16)c.x;
  b[1] = (_Float16)c.y;
  b[2] = (_Float16)c.z;
  b[3] = (_Float16)1.0f;
  b[4] = (_Float16)sq;
  b[5] = (_Float16)0.0f; b[6] = (_Float16)0.0f; b[7] = (_Float16)0.0f;
  Af16[v] = a;
  Bf16[v] = b;
  lab[v] = valid ? v : SENTINEL;
  if (v == 0) {                       // zero pad entry for hi-lane loads
    v8h z;
    #pragma unroll
    for (int e = 0; e < 8; ++e) z[e] = (_Float16)0.0f;
    Af16[MAXV] = z;
    Bf16[MAXV] = z;
  }
}

// ---------------------------------------------------------------------------
// K5: one chaotic min-label relaxation sweep over a single in-place label
// buffer. Labels are monotonically non-increasing and bounded below by the
// per-component min, so racy reads + global_atomic_min_i32 writes converge to
// the same fixed point as the reference's 64 synchronous sweeps (64 >= graph
// diameter), and async progress per sweep >= sync progress.
// 2048 waves: (i-tile, j-split). Inner loop per 16x16 tile:
//   1 b128 fragment load + 1 label load + 1 WMMA (d2 direct) + 8x(cmp,sel,min3)
// ---------------------------------------------------------------------------
__global__ void k_prop(const v8h* __restrict__ Af16, const v8h* __restrict__ Bf16,
                       int* __restrict__ lab) {
  const int w     = blockIdx.x * (blockDim.x >> 5) + (threadIdx.x >> 5);
  const int lane  = threadIdx.x & 31;
  const int lidx  = lane & 15;
  const bool lo   = (lane < 16);
  const int itile = w >> 2;
  const int split = w & (NSPLIT - 1);
  const int i0    = itile * 16;
  const int jbeg  = split * (MAXV / NSPLIT);
  const int jend  = jbeg + (MAXV / NSPLIT);

  // A fragment: lanes 0-15 carry K=0..7 in elements 0..7; all else zero
  v8h a8 = Af16[lo ? (i0 + lidx) : MAXV];
  v16h afrag;
  #pragma unroll
  for (int e = 0; e < 8; ++e) { afrag[e] = a8[e]; afrag[8 + e] = (_Float16)0.0f; }

  int runmin[8];
  #pragma unroll
  for (int v = 0; v < 8; ++v) runmin[v] = SENTINEL;

  for (int j0 = jbeg; j0 < jend; j0 += 16) {
    const int j = j0 + lidx;
    v8h b8 = Bf16[lo ? j : MAXV];
    int labj = lab[j];

    v16h bfrag;
    #pragma unroll
    for (int e = 0; e < 8; ++e) { bfrag[e] = b8[e]; bfrag[8 + e] = (_Float16)0.0f; }

    v8f d2;
    #pragma unroll
    for (int e = 0; e < 8; ++e) d2[e] = 0.0f;
    d2 = __builtin_amdgcn_wmma_f32_16x16x32_f16(
             /*neg_a=*/false, afrag, /*neg_b=*/false, bfrag,
             /*c_mod=*/(short)0, d2, /*reuse_a=*/false, /*reuse_b=*/false);

    #pragma unroll
    for (int v = 0; v < 8; ++v) {
      int cand = (d2[v] <= 4.0f) ? labj : SENTINEL;
      runmin[v] = min(runmin[v], cand);
    }
  }

  // min across the 16 lanes of each half (columns N=0..15)
  #pragma unroll
  for (int v = 0; v < 8; ++v) {
    int r = runmin[v];
    r = min(r, __shfl_xor(r, 1, 32));
    r = min(r, __shfl_xor(r, 2, 32));
    r = min(r, __shfl_xor(r, 4, 32));
    r = min(r, __shfl_xor(r, 8, 32));
    runmin[v] = r;
  }
  if (lidx == 0) {
    const int mbase = i0 + (lo ? 0 : 8);
    #pragma unroll
    for (int v = 0; v < 8; ++v) {
      atomicMin(&lab[mbase + v], runmin[v]);   // no-op for invalid rows (8192)
    }
  }
}

// ---------------------------------------------------------------------------
// K6: per-cluster segment sums
// ---------------------------------------------------------------------------
__global__ void k_stats(const float4* __restrict__ C4,
                        const float4* __restrict__ voxsum,
                        const int* __restrict__ lab,
                        float* __restrict__ csum, int* __restrict__ ccnt) {
  int v = blockIdx.x * blockDim.x + threadIdx.x;
  if (v >= MAXV) return;
  if (C4[v].w < 1e29f) {
    int s = lab[v];
    float4 vs = voxsum[v];
    atomicAdd(&csum[4 * s + 0], vs.x);
    atomicAdd(&csum[4 * s + 1], vs.y);
    atomicAdd(&csum[4 * s + 2], vs.z);
    atomicAdd(&ccnt[s], 1);
  }
}

// ---------------------------------------------------------------------------
// K7: final outputs: [labels | centers(3) | sizes | valid_clusters] as floats
// ---------------------------------------------------------------------------
__global__ void k_out(const int* __restrict__ lab, const float4* __restrict__ csum,
                      const int* __restrict__ ccnt, float* __restrict__ out) {
  int s = blockIdx.x * blockDim.x + threadIdx.x;
  if (s >= MAXV) return;
  out[s] = (float)lab[s];
  float size = (float)ccnt[s] * (float)MAXP;
  float den  = fmaxf(size, 1.0f);
  float4 cs  = csum[s];
  out[MAXV + 3 * s + 0] = cs.x / den;
  out[MAXV + 3 * s + 1] = cs.y / den;
  out[MAXV + 3 * s + 2] = cs.z / den;
  out[MAXV * 4 + s] = size;
  out[MAXV * 5 + s] = (size >= 50.0f && size <= 5000.0f) ? 1.0f : 0.0f;
}

// ---------------------------------------------------------------------------
extern "C" void kernel_launch(void* const* d_in, const int* in_sizes, int n_in,
                              void* d_out, int out_size, void* d_ws, size_t ws_size,
                              hipStream_t stream) {
  const float4* pts = (const float4*)d_in[0];
  char* ws = (char*)d_ws;

  size_t off = 0;
  auto alloc = [&](size_t bytes) -> char* {
    char* p = ws + off;
    off += (bytes + 255) & ~(size_t)255;
    return p;
  };
  float4* C4        = (float4*)alloc(MAXV * 16);        // x,y,z,|C|^2 (1e30 = invalid)
  float4* voxsum    = (float4*)alloc(MAXV * 16);        // capped xyz sums
  float4* csum      = (float4*)alloc(MAXV * 16);        // cluster centroid sums
  v8h*    Af16      = (v8h*)alloc((MAXV + 1) * 16);     // A fragments (+zero pad)
  v8h*    Bf16      = (v8h*)alloc((MAXV + 1) * 16);     // B fragments (+zero pad)
  int*    cnt_cells = (int*)alloc(MAXV * 4);
  int*    cellToVid = (int*)alloc(MAXV * 4);
  int*    fill      = (int*)alloc(MAXV * 4);
  int*    lab       = (int*)alloc(MAXV * 4);            // single in-place label buffer
  int*    ccnt      = (int*)alloc(MAXV * 4);
  float*  out       = (float*)d_out;

  k_init<<<MAXV / 256, 256, 0, stream>>>(C4, voxsum, csum, cnt_cells, cellToVid, fill, ccnt);
  k_count<<<(NPTS + 255) / 256, 256, 0, stream>>>(pts, cnt_cells);
  k_scan<<<1, 1024, 0, stream>>>(cnt_cells, cellToVid, C4);
  k_accum<<<(NPTS + 255) / 256, 256, 0, stream>>>(pts, cellToVid, fill, (float*)voxsum);
  k_prep<<<MAXV / 256, 256, 0, stream>>>(C4, Af16, Bf16, lab);

  for (int it = 0; it < PROP_ITERS; ++it) {  // 2048 waves: 256 blocks x 8 waves
    k_prop<<<256, 256, 0, stream>>>(Af16, Bf16, lab);
  }

  k_stats<<<MAXV / 256, 256, 0, stream>>>(C4, voxsum, lab, (float*)csum, ccnt);
  k_out<<<MAXV / 256, 256, 0, stream>>>(lab, csum, ccnt, out);
}
